// CausalSelfAttention_83030307766301
// MI455X (gfx1250) — matrix operand order repaired and checked by reading, not compiled, not run on hardware
//
#include <hip/hip_runtime.h>
#include <cstdint>
#include <cstddef>

#define BATCH   2
#define SEQ     2048
#define DMODEL  1024
#define H_HEADS 16
#define HD      64

typedef __attribute__((ext_vector_type(16))) __bf16   v16bf;
typedef __attribute__((ext_vector_type(8)))  __bf16   v8bf;
typedef __attribute__((ext_vector_type(4)))  __bf16   v4bf;
typedef __attribute__((ext_vector_type(8)))  float    v8f;
typedef __attribute__((ext_vector_type(4)))  uint32_t v4ui;
typedef __attribute__((ext_vector_type(8)))  int      v8i;
typedef __attribute__((ext_vector_type(4)))  int      v4i;

#if defined(__has_builtin)
#if __has_builtin(__builtin_amdgcn_tensor_load_to_lds) && \
    __has_builtin(__builtin_amdgcn_s_wait_tensorcnt)
#define USE_TDM 1
#endif
#endif
#ifndef USE_TDM
#define USE_TDM 0
#endif

static __device__ __forceinline__ v16bf pack16(v8bf lo, v8bf hi) {
  union { v16bf v; v8bf h[2]; } u;
  u.h[0] = lo; u.h[1] = hi;
  return u.v;
}

static __device__ __forceinline__ v8f wmma_bf16(v16bf a, v16bf b, v8f c) {
  return __builtin_amdgcn_wmma_f32_16x16x32_bf16(
      false, a, false, b, (short)0, c, false, false);
}

#if USE_TDM
// One 2-D TDM descriptor: copy a [rows x 32] bf16 tile (row stride = ldk
// elements) from global to LDS, packed row-major [rows][32].
// amdgpu-toolchain (clang-23) 6-arg builtin:
//   (uint32x4 g0, int32x8 g1, int32x4 g2, int32x4 g3, int32x8 extra, i32 cpol)
static __device__ __forceinline__ void tdm_load_tile(
    const __bf16* gsrc, uint32_t lds_off, int rows, int ldk)
{
  const uint64_t ga = (uint64_t)(uintptr_t)gsrc;
  v4ui g0;
  g0[0] = 1u;                                                  // count=1
  g0[1] = lds_off;                                             // lds_addr
  g0[2] = (uint32_t)ga;                                        // global[31:0]
  g0[3] = (uint32_t)((ga >> 32) & 0x01FFFFFFu) | 0x80000000u;  // global[56:32] | type=2
  v8i g1;
  g1[0] = 0x00010000;                                    // data_size = 2 bytes
  g1[1] = (int)(((uint32_t)ldk & 0xFFFFu) << 16);        // tensor_dim0[15:0]
  g1[2] = (int)(((uint32_t)ldk >> 16) & 0xFFFFu);        // tensor_dim0[31:16], td1[15:0]=0
  g1[3] = (int)(0x10u | (32u << 16));                    // td1[31:16] (=1M rows), tile_dim0=32
  g1[4] = rows;                                          // tile_dim1 = rows, tile_dim2=0
  g1[5] = ldk;                                           // tensor_dim0_stride[31:0]
  g1[6] = 0;                                             // stride0[47:32], stride1[15:0]
  g1[7] = 0;                                             // stride1[47:16]
  const v4i z4 = { 0, 0, 0, 0 };                         // 2-D: groups 2/3 unused
  const v8i z8 = { 0, 0, 0, 0, 0, 0, 0, 0 };
  __builtin_amdgcn_tensor_load_to_lds(g0, g1, z4, z4, z8, 0);
}
#endif

// ---------------------------------------------------------------------------
// f32 -> bf16 conversion (vectorized 4-wide)
// ---------------------------------------------------------------------------
__global__ __launch_bounds__(256) void cvt_f32_to_bf16(
    const float* __restrict__ in, __bf16* __restrict__ out, int n4)
{
  int i = blockIdx.x * blockDim.x + threadIdx.x;
  if (i < n4) {
    float4 v = ((const float4*)in)[i];
    v4bf o = { (__bf16)v.x, (__bf16)v.y, (__bf16)v.z, (__bf16)v.w };
    *(v4bf*)(out + (size_t)i * 4) = o;
  }
}

// ---------------------------------------------------------------------------
// Tiled f32[K,N] -> bf16[N,K] transpose+convert (one-time weight prep)
// ---------------------------------------------------------------------------
__global__ __launch_bounds__(256) void transpose_cvt_bf16(
    const float* __restrict__ in, __bf16* __restrict__ out, int K, int N)
{
  __shared__ float tile[32][33];
  const int tx = threadIdx.x;            // 0..31
  const int ty = threadIdx.y;            // 0..7
  const int kbase = blockIdx.y * 32;
  const int nbase = blockIdx.x * 32;
#pragma unroll
  for (int i = ty; i < 32; i += 8)
    tile[i][tx] = in[(size_t)(kbase + i) * N + nbase + tx];
  __syncthreads();
#pragma unroll
  for (int i = ty; i < 32; i += 8)
    out[(size_t)(nbase + i) * K + kbase + tx] = (__bf16)tile[tx][i];
}

// ---------------------------------------------------------------------------
// Tiled bf16 GEMM:  C = A[M,K] @ Wt[N,K]^T + bias[N]
//   (Wt is the pre-transposed weight: row n = column n of W)
//   MODE 0: f32 out[M,N];  MODE 1: scatter q/k ([S,HD]) + transposed vT ([HD,S])
// Block: 128 thr (4 waves). Block tile 128x64, K-step 32.
// Wave w: rows [w*32, w*32+32) x 64 cols -> 8 WMMAs / K-step.
// A and B tiles staged by the Tensor Data Mover (waves 0/1), TENSORcnt-synced.
// ---------------------------------------------------------------------------
template<int MODE>
__global__ __launch_bounds__(128) void gemm_bf16_wmma(
    const __bf16* __restrict__ A,
    const __bf16* __restrict__ Wt,
    const float*  __restrict__ bias,
    float*   __restrict__ outF,
    __bf16*  __restrict__ qbuf,
    __bf16*  __restrict__ kbuf,
    __bf16*  __restrict__ vTbuf,
    int M, int N, int K)
{
  __shared__ __bf16 As[128 * 32];   // [m][k]
  __shared__ __bf16 Bs[64 * 32];    // [n][k]

  const int tid  = threadIdx.x;
  const int wave = tid >> 5;
  const int lane = tid & 31;
  const int l16  = lane & 15;
  const int lh   = lane >> 4;
  const int bm   = blockIdx.y * 128;
  const int bn   = blockIdx.x * 64;

  v8f acc[2][4] = {};

  for (int k0 = 0; k0 < K; k0 += 32) {
#if USE_TDM
    if (wave == 0)
      tdm_load_tile(A  + (size_t)bm * K + k0, (uint32_t)(uintptr_t)As, 128, K);
    if (wave == 1)
      tdm_load_tile(Wt + (size_t)bn * K + k0, (uint32_t)(uintptr_t)Bs,  64, K);
    __builtin_amdgcn_s_wait_tensorcnt((short)0);
#else
#pragma unroll
    for (int i = 0; i < 4; ++i) {              // A: 128x32 = 512 16B chunks
      int c = i * 128 + tid;
      int row = c >> 2, seg = c & 3;
      *(uint4*)(As + row * 32 + seg * 8) =
          *(const uint4*)(A + (size_t)(bm + row) * K + k0 + seg * 8);
    }
#pragma unroll
    for (int i = 0; i < 2; ++i) {              // B: 64x32 = 256 16B chunks
      int c = i * 128 + tid;
      int row = c >> 2, seg = c & 3;
      *(uint4*)(Bs + row * 32 + seg * 8) =
          *(const uint4*)(Wt + (size_t)(bn + row) * K + k0 + seg * 8);
    }
    if (k0 + 32 < K) {
      __builtin_prefetch(A  + (size_t)(bm + (tid >> 2)) * K + k0 + 32, 0, 1);
      __builtin_prefetch(Wt + (size_t)(bn + (tid >> 1)) * K + k0 + 32, 0, 1);
    }
#endif
    __syncthreads();

    v16bf afrag[2];
#pragma unroll
    for (int mi = 0; mi < 2; ++mi) {
      const int arow = wave * 32 + mi * 16 + l16;
      v8bf lo = *(const v8bf*)(As + arow * 32 + lh * 8);
      v8bf hi = *(const v8bf*)(As + arow * 32 + 16 + lh * 8);
      afrag[mi] = pack16(lo, hi);
    }
#pragma unroll
    for (int n = 0; n < 4; ++n) {
      const int brow = n * 16 + l16;
      v8bf blo = *(const v8bf*)(Bs + brow * 32 + lh * 8);
      v8bf bhi = *(const v8bf*)(Bs + brow * 32 + 16 + lh * 8);
      v16bf bfrag = pack16(blo, bhi);
      acc[0][n] = wmma_bf16(afrag[0], bfrag, acc[0][n]);
      acc[1][n] = wmma_bf16(afrag[1], bfrag, acc[1][n]);
    }
    __syncthreads();
  }

  // Epilogue. C layout: lane holds col N=l16, reg r holds row M = r + 8*lh.
#pragma unroll
  for (int mi = 0; mi < 2; ++mi) {
#pragma unroll
    for (int n = 0; n < 4; ++n) {
      const int col = bn + n * 16 + l16;
      const float bv = bias[col];
#pragma unroll
      for (int r = 0; r < 8; ++r) {
        const int row = bm + wave * 32 + mi * 16 + r + 8 * lh;
        const float val = acc[mi][n][r] + bv;
        if (MODE == 0) {
          outF[(size_t)row * N + col] = val;
        } else {
          const int which = col >> 10;       // 0=q 1=k 2=v
          const int d  = col & 1023;
          const int h  = d >> 6;
          const int hd = d & 63;
          const int b  = row >> 11;
          const int s  = row & 2047;
          const size_t head = (size_t)(b * H_HEADS + h);
          const __bf16 bvv = (__bf16)val;
          if (which == 0)      qbuf [(head * SEQ + s) * HD + hd] = bvv;
          else if (which == 1) kbuf [(head * SEQ + s) * HD + hd] = bvv;
          else                 vTbuf[(head * HD + hd) * SEQ + s] = bvv;
        }
      }
    }
  }
}

// ---------------------------------------------------------------------------
// Flash-attention forward. One wave per 16-query tile (4 waves / block).
// ---------------------------------------------------------------------------
__global__ __launch_bounds__(128) void attn_fwd_wmma(
    const __bf16* __restrict__ q,    // [B*H][S][HD]
    const __bf16* __restrict__ k,    // [B*H][S][HD]
    const __bf16* __restrict__ vT,   // [B*H][HD][S]
    __bf16* __restrict__ ctx)        // [B][S][DMODEL]
{
  __shared__ __bf16 p_lds[4][16 * 32];

  const int tid  = threadIdx.x;
  const int wave = tid >> 5;
  const int lane = tid & 31;
  const int l16  = lane & 15;
  const int lh   = lane >> 4;

  const int qtile = blockIdx.x * 4 + wave;
  const int h     = blockIdx.y;
  const int b     = blockIdx.z;
  const int qbase = qtile * 16;

  const size_t head = (size_t)(b * H_HEADS + h);
  const __bf16* qh = q  + head * (size_t)SEQ * HD;
  const __bf16* kh = k  + head * (size_t)SEQ * HD;
  const __bf16* vh = vT + head * (size_t)HD * SEQ;

  const int qrow = qbase + l16;
  v16bf qfrag[2];
#pragma unroll
  for (int kk = 0; kk < 2; ++kk) {
    v8bf lo = *(const v8bf*)(qh + (size_t)qrow * HD + kk * 32 + lh * 8);
    v8bf hi = *(const v8bf*)(qh + (size_t)qrow * HD + kk * 32 + 16 + lh * 8);
    qfrag[kk] = pack16(lo, hi);
  }

  v8f acc[4] = {};
  float m_r[8], l_r[8];
#pragma unroll
  for (int r = 0; r < 8; ++r) { m_r[r] = -__builtin_inff(); l_r[r] = 0.0f; }

  const float scale = 0.125f;
  const int kend = qbase + 16;

  for (int kb = 0; kb < kend; kb += 32) {
    v8f sc[2] = {};
#pragma unroll
    for (int nn = 0; nn < 2; ++nn) {
      const int kcol = kb + nn * 16 + l16;
#pragma unroll
      for (int kk = 0; kk < 2; ++kk) {
        v8bf lo = *(const v8bf*)(kh + (size_t)kcol * HD + kk * 32 + lh * 8);
        v8bf hi = *(const v8bf*)(kh + (size_t)kcol * HD + kk * 32 + 16 + lh * 8);
        sc[nn] = wmma_bf16(qfrag[kk], pack16(lo, hi), sc[nn]);
      }
    }

#pragma unroll
    for (int r = 0; r < 8; ++r) {
      const int qidx = qbase + r + 8 * lh;
      float s0 = sc[0][r] * scale;
      float s1 = sc[1][r] * scale;
      if (kb + l16      > qidx) s0 = -__builtin_inff();
      if (kb + 16 + l16 > qidx) s1 = -__builtin_inff();
      float mx = fmaxf(s0, s1);
      mx = fmaxf(mx, __shfl_xor(mx, 1, 16));
      mx = fmaxf(mx, __shfl_xor(mx, 2, 16));
      mx = fmaxf(mx, __shfl_xor(mx, 4, 16));
      mx = fmaxf(mx, __shfl_xor(mx, 8, 16));
      const float mnew = fmaxf(m_r[r], mx);
      const float p0 = __expf(s0 - mnew);
      const float p1 = __expf(s1 - mnew);
      float ps = p0 + p1;
      ps += __shfl_xor(ps, 1, 16);
      ps += __shfl_xor(ps, 2, 16);
      ps += __shfl_xor(ps, 4, 16);
      ps += __shfl_xor(ps, 8, 16);
      const float alpha = __expf(m_r[r] - mnew);
      l_r[r] = l_r[r] * alpha + ps;
      m_r[r] = mnew;
#pragma unroll
      for (int n = 0; n < 4; ++n) acc[n][r] *= alpha;
      p_lds[wave][(r + 8 * lh) * 32 + l16]      = (__bf16)p0;
      p_lds[wave][(r + 8 * lh) * 32 + 16 + l16] = (__bf16)p1;
    }

    v8bf plo = *(const v8bf*)(&p_lds[wave][l16 * 32 + lh * 8]);
    v8bf phi = *(const v8bf*)(&p_lds[wave][l16 * 32 + 16 + lh * 8]);
    v16bf pfrag = pack16(plo, phi);
#pragma unroll
    for (int n = 0; n < 4; ++n) {
      const __bf16* vp = vh + (size_t)(n * 16 + l16) * SEQ + kb + 8 * lh;
      v8bf lo = *(const v8bf*)vp;
      v8bf hi = *(const v8bf*)(vp + 16);
      acc[n] = wmma_bf16(pfrag, pack16(lo, hi), acc[n]);
    }
  }

#pragma unroll
  for (int r = 0; r < 8; ++r) {
    const float inv = 1.0f / l_r[r];
    const int s = qbase + r + 8 * lh;
#pragma unroll
    for (int n = 0; n < 4; ++n) {
      const int d = h * HD + n * 16 + l16;
      ctx[((size_t)(b * SEQ + s)) * DMODEL + d] = (__bf16)(acc[n][r] * inv);
    }
  }
}

// ---------------------------------------------------------------------------
extern "C" void kernel_launch(void* const* d_in, const int* in_sizes, int n_in,
                              void* d_out, int out_size, void* d_ws, size_t ws_size,
                              hipStream_t stream)
{
  (void)in_sizes; (void)n_in; (void)out_size; (void)ws_size;
  const float* x      = (const float*)d_in[0];
  const float* w_attn = (const float*)d_in[1];
  const float* b_attn = (const float*)d_in[2];
  const float* w_proj = (const float*)d_in[3];
  const float* b_proj = (const float*)d_in[4];
  float* out = (float*)d_out;

  const int M = BATCH * SEQ;           // 4096
  char* p = (char*)d_ws;
  __bf16* xb   = (__bf16*)p; p += (size_t)M * DMODEL * 2;
  __bf16* wabT = (__bf16*)p; p += (size_t)3 * DMODEL * DMODEL * 2;  // [3D][D]
  __bf16* wpbT = (__bf16*)p; p += (size_t)DMODEL * DMODEL * 2;      // [D][D]
  __bf16* qb   = (__bf16*)p; p += (size_t)BATCH * H_HEADS * SEQ * HD * 2;
  __bf16* kb   = (__bf16*)p; p += (size_t)BATCH * H_HEADS * SEQ * HD * 2;
  __bf16* vTb  = (__bf16*)p; p += (size_t)BATCH * H_HEADS * HD * SEQ * 2;
  __bf16* ctx  = (__bf16*)p; p += (size_t)M * DMODEL * 2;

  // 1) activations -> bf16; weights -> transposed bf16 [N][K]
  {
    int n4 = M * DMODEL / 4;
    cvt_f32_to_bf16<<<dim3((n4 + 255) / 256), 256, 0, stream>>>(x, xb, n4);
    transpose_cvt_bf16<<<dim3(3 * DMODEL / 32, DMODEL / 32), dim3(32, 8), 0, stream>>>(
        w_attn, wabT, DMODEL, 3 * DMODEL);
    transpose_cvt_bf16<<<dim3(DMODEL / 32, DMODEL / 32), dim3(32, 8), 0, stream>>>(
        w_proj, wpbT, DMODEL, DMODEL);
  }

  // 2) QKV GEMM with fused head-split / v-transpose scatter
  gemm_bf16_wmma<1><<<dim3(3 * DMODEL / 64, M / 128), 128, 0, stream>>>(
      xb, wabT, b_attn, nullptr, qb, kb, vTb, M, 3 * DMODEL, DMODEL);

  // 3) causal flash attention
  attn_fwd_wmma<<<dim3(SEQ / 16 / 4, H_HEADS, BATCH), 128, 0, stream>>>(
      qb, kb, vTb, ctx);

  // 4) output projection (f32 result)
  gemm_bf16_wmma<0><<<dim3(DMODEL / 64, M / 128), 128, 0, stream>>>(
      ctx, wpbT, b_proj, out, nullptr, nullptr, nullptr, M, DMODEL, DMODEL);
}
